// RNN_52295521796124
// MI455X (gfx1250) — compile-verified, matrix operand order checked by reading, not run on hardware
//
#include <hip/hip_runtime.h>

#define T_STEPS 512
#define BATCH   64
#define DIN     1024
#define HDIM    1024
#define DOUT    1024
#define KP      256   // K-panel staged in LDS by the step kernel
#define NPANEL  (HDIM / KP)

typedef __attribute__((ext_vector_type(16))) __bf16 v16bf;
typedef __attribute__((ext_vector_type(8)))  __bf16 v8bf;
typedef __attribute__((ext_vector_type(8)))  float  v8f;
typedef __attribute__((ext_vector_type(4)))  float  v4f;

// D = A(16x32 bf16) * B(32x16 bf16) + C(16x16 f32), wave32.
__device__ __forceinline__ v8f wmma_bf16(v16bf a, v16bf b, v8f c) {
  return __builtin_amdgcn_wmma_f32_16x16x32_bf16(
      /*neg_a=*/false, a, /*neg_b=*/false, b,
      /*c_mod=*/(short)0, c, /*reuse_a=*/false, /*reuse_b=*/false);
}

// 16-bit A/B VGPR layout: lane l owns row (l&15); half-wave g = l>>4 holds
// K = {k0+g*8..+7} in elems 0..7 and K = {k0+16+g*8..+7} in 8..15.
// Works for both global and LDS row pointers (16B-aligned chunks).
template <typename P>
__device__ __forceinline__ v16bf load_bf16_op(P row, int k0, int g) {
  const v8bf lo = *(const v8bf*)(row + k0 + g * 8);
  const v8bf hi = *(const v8bf*)(row + k0 + 16 + g * 8);
  v16bf r;
#pragma unroll
  for (int i = 0; i < 8; ++i) { r[i] = lo[i]; r[i + 8] = hi[i]; }
  return r;
}

// Same pattern, fp32 source converted to bf16 in-register (cvt co-issues w/ WMMA).
__device__ __forceinline__ v16bf load_f32_op(const float* __restrict__ row,
                                             int k0, int g) {
  const v4f a0 = *(const v4f*)(row + k0 + g * 8);
  const v4f a1 = *(const v4f*)(row + k0 + g * 8 + 4);
  const v4f b0 = *(const v4f*)(row + k0 + 16 + g * 8);
  const v4f b1 = *(const v4f*)(row + k0 + 16 + g * 8 + 4);
  v16bf r;
#pragma unroll
  for (int i = 0; i < 4; ++i) {
    r[i]      = (__bf16)a0[i];
    r[i + 4]  = (__bf16)a1[i];
    r[i + 8]  = (__bf16)b0[i];
    r[i + 12] = (__bf16)b1[i];
  }
  return r;
}

// dst[C][R] (bf16) = transpose(src[R][C] (f32)).
__global__ void transpose_bf16_kernel(const float* __restrict__ src,
                                      __bf16* __restrict__ dst, int R, int C) {
  int idx = blockIdx.x * blockDim.x + threadIdx.x;
  if (idx >= R * C) return;
  int c = idx / R, r = idx - c * R;
  dst[idx] = (__bf16)src[(size_t)r * C + c];
}

__global__ void cvt_bf16_kernel(const float* __restrict__ src,
                                __bf16* __restrict__ dst, int n) {
  int i = blockIdx.x * blockDim.x + threadIdx.x;
  if (i < n) dst[i] = (__bf16)src[i];
}

// Phase 1: xW[32768][1024] = x @ W_h + b_h. Wave tile 32x64 (8 wmma / K-step).
__global__ __launch_bounds__(256) void gemm_xw_kernel(
    const float* __restrict__ x, const __bf16* __restrict__ WhT,
    const float* __restrict__ b_h, float* __restrict__ xW) {
  const int lane = threadIdx.x & 31;
  const int g  = lane >> 4;
  const int lr = lane & 15;
  const int wave = blockIdx.x * 8 + (threadIdx.x >> 5);
  if (wave >= (T_STEPS * BATCH / 32) * (HDIM / 64)) return;
  const int m2 = wave >> 4;
  const int n4 = wave & 15;

  const float* arow0 = x + (size_t)(m2 * 32 + lr) * DIN;
  const float* arow1 = arow0 + (size_t)16 * DIN;
  const __bf16* brow[4];
#pragma unroll
  for (int j = 0; j < 4; ++j)
    brow[j] = WhT + (size_t)(n4 * 64 + j * 16 + lr) * DIN;

  v8f c[2][4] = {};
  for (int k0 = 0; k0 < DIN; k0 += 32) {
    v16bf a0 = load_f32_op(arow0, k0, g);
    v16bf a1 = load_f32_op(arow1, k0, g);
    v16bf b0 = load_bf16_op(brow[0], k0, g);
    v16bf b1 = load_bf16_op(brow[1], k0, g);
    v16bf b2 = load_bf16_op(brow[2], k0, g);
    v16bf b3 = load_bf16_op(brow[3], k0, g);
    c[0][0] = wmma_bf16(a0, b0, c[0][0]);
    c[0][1] = wmma_bf16(a0, b1, c[0][1]);
    c[0][2] = wmma_bf16(a0, b2, c[0][2]);
    c[0][3] = wmma_bf16(a0, b3, c[0][3]);
    c[1][0] = wmma_bf16(a1, b0, c[1][0]);
    c[1][1] = wmma_bf16(a1, b1, c[1][1]);
    c[1][2] = wmma_bf16(a1, b2, c[1][2]);
    c[1][3] = wmma_bf16(a1, b3, c[1][3]);
  }
#pragma unroll
  for (int mt = 0; mt < 2; ++mt) {
    const int mbase = m2 * 32 + mt * 16 + 8 * g;  // + r; C: VGPR r -> M=r+8g, N=lr
#pragma unroll
    for (int nt = 0; nt < 4; ++nt) {
      const int n = n4 * 64 + nt * 16 + lr;
      const float bias = b_h[n];
      float* orow = xW + (size_t)mbase * HDIM + n;
#pragma unroll
      for (int r = 0; r < 8; ++r)
        orow[(size_t)r * HDIM] = c[mt][nt][r] + bias;
    }
  }
}

// Phase 2 (serial, latency-critical): h_{s+1} = tanh(xW_s + h_s @ U_h).
// 8 blocks x 8 waves; wave owns a 64x16 strip. h_s staged through a
// double-buffered LDS panel pipeline: next panel's global loads are issued
// before the current panel's 32 WMMAs, hiding L2 latency; one barrier/panel.
__global__ __launch_bounds__(256) void rnn_h_step_kernel(
    const __bf16* __restrict__ h_in,   // [64][1024] bf16 (h_s)
    const __bf16* __restrict__ UT,     // [1024][1024] bf16, [N][K]
    const float* __restrict__ xW_t,    // [64][1024] (b_h folded in)
    __bf16* __restrict__ h_out,        // [64][1024] bf16 -> Hst[s]
    float* __restrict__ h_out_f32,     // final-state fp32 (last step only)
    int write_f32) {
  __shared__ __bf16 hsh[2][BATCH * KP];   // 2 x 32 KB panels of h_s
  const int tid  = threadIdx.x;
  const int lane = tid & 31;
  const int g  = lane >> 4;
  const int lr = lane & 15;
  const int ntile = blockIdx.x * 8 + (tid >> 5);  // 0..63
  const int n = ntile * 16 + lr;
  const __bf16* brow = UT + (size_t)n * HDIM;

  const int NCH = (BATCH * KP / 8) / 256;  // 16B chunks per thread per panel (=8)
  uint4 stage[(BATCH * KP / 8) / 256];

  // Prologue: stage panel 0 into buffer 0.
  {
    const uint4* src = (const uint4*)h_in;
#pragma unroll
    for (int i = 0; i < NCH; ++i) {
      const int cidx = tid + i * 256;
      stage[i] = src[(size_t)(cidx >> 5) * (HDIM / 8) + (cidx & 31)];
    }
    uint4* dst = (uint4*)hsh[0];
#pragma unroll
    for (int i = 0; i < NCH; ++i) dst[tid + i * 256] = stage[i];
  }
  __syncthreads();

  v8f c[4] = {};
  for (int p = 0; p < NPANEL; ++p) {
    // Issue next panel's global loads now; they drain behind the WMMAs below.
    if (p + 1 < NPANEL) {
      const uint4* src = (const uint4*)(h_in + (p + 1) * KP);
#pragma unroll
      for (int i = 0; i < NCH; ++i) {
        const int cidx = tid + i * 256;
        stage[i] = src[(size_t)(cidx >> 5) * (HDIM / 8) + (cidx & 31)];
      }
    }
    const __bf16* hbuf = hsh[p & 1];
#pragma unroll
    for (int k0 = 0; k0 < KP; k0 += 32) {
      v16bf b = load_bf16_op(brow, p * KP + k0, g);
#pragma unroll
      for (int mt = 0; mt < 4; ++mt) {
        v16bf a = load_bf16_op(hbuf + (mt * 16 + lr) * KP, k0, g);
        c[mt] = wmma_bf16(a, b, c[mt]);
      }
    }
    if (p + 1 < NPANEL) {
      uint4* dst = (uint4*)hsh[(p + 1) & 1];
#pragma unroll
      for (int i = 0; i < NCH; ++i) dst[tid + i * 256] = stage[i];
      __syncthreads();  // next panel visible; old panel no longer read
    }
  }

#pragma unroll
  for (int mt = 0; mt < 4; ++mt) {
    const int mbase = mt * 16 + 8 * g;
#pragma unroll
    for (int r = 0; r < 8; ++r) {
      const size_t idx = (size_t)(mbase + r) * HDIM + n;
      const float v = tanhf(xW_t[idx] + c[mt][r]);
      h_out[idx] = (__bf16)v;
      if (write_f32) h_out_f32[idx] = v;
    }
  }
}

// Phase 3: Y[32768][1024] = tanh(Hst @ W_y + b_y). Fully parallel big GEMM.
__global__ __launch_bounds__(256) void gemm_y_kernel(
    const __bf16* __restrict__ Hst, const __bf16* __restrict__ WyT,
    const float* __restrict__ b_y, float* __restrict__ y) {
  const int lane = threadIdx.x & 31;
  const int g  = lane >> 4;
  const int lr = lane & 15;
  const int wave = blockIdx.x * 8 + (threadIdx.x >> 5);
  if (wave >= (T_STEPS * BATCH / 32) * (DOUT / 64)) return;
  const int m2 = wave >> 4;
  const int n4 = wave & 15;

  const __bf16* arow0 = Hst + (size_t)(m2 * 32 + lr) * HDIM;
  const __bf16* arow1 = arow0 + (size_t)16 * HDIM;
  const __bf16* brow[4];
#pragma unroll
  for (int j = 0; j < 4; ++j)
    brow[j] = WyT + (size_t)(n4 * 64 + j * 16 + lr) * HDIM;

  v8f c[2][4] = {};
  for (int k0 = 0; k0 < HDIM; k0 += 32) {
    v16bf a0 = load_bf16_op(arow0, k0, g);
    v16bf a1 = load_bf16_op(arow1, k0, g);
    v16bf b0 = load_bf16_op(brow[0], k0, g);
    v16bf b1 = load_bf16_op(brow[1], k0, g);
    v16bf b2 = load_bf16_op(brow[2], k0, g);
    v16bf b3 = load_bf16_op(brow[3], k0, g);
    c[0][0] = wmma_bf16(a0, b0, c[0][0]);
    c[0][1] = wmma_bf16(a0, b1, c[0][1]);
    c[0][2] = wmma_bf16(a0, b2, c[0][2]);
    c[0][3] = wmma_bf16(a0, b3, c[0][3]);
    c[1][0] = wmma_bf16(a1, b0, c[1][0]);
    c[1][1] = wmma_bf16(a1, b1, c[1][1]);
    c[1][2] = wmma_bf16(a1, b2, c[1][2]);
    c[1][3] = wmma_bf16(a1, b3, c[1][3]);
  }
#pragma unroll
  for (int mt = 0; mt < 2; ++mt) {
    const int mbase = m2 * 32 + mt * 16 + 8 * g;
#pragma unroll
    for (int nt = 0; nt < 4; ++nt) {
      const int n = n4 * 64 + nt * 16 + lr;
      const float bias = b_y[n];
      float* orow = y + (size_t)mbase * DOUT + n;
#pragma unroll
      for (int r = 0; r < 8; ++r)
        orow[(size_t)r * DOUT] = tanhf(c[mt][nt][r] + bias);
    }
  }
}

extern "C" void kernel_launch(void* const* d_in, const int* in_sizes, int n_in,
                              void* d_out, int out_size, void* d_ws,
                              size_t ws_size, hipStream_t stream) {
  (void)in_sizes; (void)n_in; (void)out_size; (void)ws_size;
  const float* x   = (const float*)d_in[0];
  const float* h0  = (const float*)d_in[1];
  const float* W_h = (const float*)d_in[2];
  const float* U_h = (const float*)d_in[3];
  const float* W_y = (const float*)d_in[4];
  const float* b_h = (const float*)d_in[5];
  const float* b_y = (const float*)d_in[6];
  float* out = (float*)d_out;

  char* ws = (char*)d_ws;
  size_t off = 0;
  float*  xW   = (float*)(ws + off);  off += (size_t)T_STEPS * BATCH * HDIM * 4;
  __bf16* Hst  = (__bf16*)(ws + off); off += (size_t)T_STEPS * BATCH * HDIM * 2;
  __bf16* WhT  = (__bf16*)(ws + off); off += (size_t)DIN  * HDIM * 2;
  __bf16* UT   = (__bf16*)(ws + off); off += (size_t)HDIM * HDIM * 2;
  __bf16* WyT  = (__bf16*)(ws + off); off += (size_t)HDIM * DOUT * 2;
  __bf16* h0b  = (__bf16*)(ws + off); off += (size_t)BATCH * HDIM * 2;
  float*  hf32 = (float*)(ws + off);  off += (size_t)BATCH * HDIM * 4;

  // One-time prep: bf16 transposed weights ([N][K]) + bf16 h0.
  const int TE = DIN * HDIM;
  transpose_bf16_kernel<<<(TE + 255) / 256, 256, 0, stream>>>(W_h, WhT, DIN, HDIM);
  transpose_bf16_kernel<<<(TE + 255) / 256, 256, 0, stream>>>(U_h, UT, HDIM, HDIM);
  transpose_bf16_kernel<<<(TE + 255) / 256, 256, 0, stream>>>(W_y, WyT, HDIM, DOUT);
  cvt_bf16_kernel<<<(BATCH * HDIM + 255) / 256, 256, 0, stream>>>(h0, h0b, BATCH * HDIM);

  // Phase 1: parallel input-projection GEMM (b_h folded in).
  gemm_xw_kernel<<<2048, 256, 0, stream>>>(x, WhT, b_h, xW);

  // Phase 2: serial recurrence; state log Hst[s] = h_{s+1} (bf16).
  const size_t BH = (size_t)BATCH * HDIM;
  for (int s = 0; s < T_STEPS; ++s) {
    const __bf16* hin = (s == 0) ? h0b : (Hst + (size_t)(s - 1) * BH);
    rnn_h_step_kernel<<<8, 256, 0, stream>>>(
        hin, UT, xW + (size_t)s * BH, Hst + (size_t)s * BH, hf32,
        (s == T_STEPS - 1) ? 1 : 0);
  }

  // Phase 3: all y_t in one throughput-bound GEMM straight into d_out.
  gemm_y_kernel<<<2048, 256, 0, stream>>>(Hst, WyT, b_y, out);

  // h_final (fp32) follows y in d_out.
  hipMemcpyAsync(out + (size_t)T_STEPS * BATCH * DOUT, hf32,
                 (size_t)BATCH * HDIM * 4, hipMemcpyDeviceToDevice, stream);
}
